// DeformableTransformerEncoderLayer_12335146074357
// MI455X (gfx1250) — matrix operand order repaired and checked by reading, not compiled, not run on hardware
//
#include <hip/hip_runtime.h>
#include <stdint.h>

#define D_ 256
#define NH_ 8
#define DH_ 32
#define NL_ 4
#define NP_ 4
#define DFF_ 1024
#define B_ 2
#define S_TOT 21760
#define BS_ (B_ * S_TOT)   // 43520

typedef __attribute__((ext_vector_type(16))) _Float16 v16h;
typedef __attribute__((ext_vector_type(8)))  float    v8f;

union Frag16 {
    uint32_t u[8];
    v16h     h;
};

union PackPair {
    _Float16 hh[2];
    uint32_t uu;
};

// ---------------------------------------------------------------------------
// Pack weight matrix W[K,N] (f32 row-major) into WMMA B-fragment layout.
// B 32x16 f16 fragment (contiguous k order chosen here, same as A pack): lanes 0-15 hold K=0..15, lanes 16-31 hold
// K=16..31; VGPR i packs rows 2i,2i+1 of that half. Stored per tile as
// [lane][8 dwords] so fragment loads are two b128 per lane.
// ---------------------------------------------------------------------------
__global__ void pack_b_kernel(const float* __restrict__ W, uint32_t* __restrict__ dst,
                              int K, int N) {
    int t = blockIdx.x * blockDim.x + threadIdx.x;
    int total = (K >> 5) * (N >> 4) * 256;
    if (t >= total) return;
    int lane = t & 31;
    int i    = (t >> 5) & 7;
    int tile = t >> 8;
    int ntiles = N >> 4;
    int nt = tile % ntiles;
    int kt = tile / ntiles;
    int col  = (nt << 4) + (lane & 15);
    int krow = (kt << 5) + ((lane >> 4) << 4) + (i << 1);
    PackPair p;
    p.hh[0] = (_Float16)W[(size_t)krow * N + col];
    p.hh[1] = (_Float16)W[(size_t)(krow + 1) * N + col];
    dst[(size_t)tile * 256 + lane * 8 + i] = p.uu;
}

// ---------------------------------------------------------------------------
// Pack activation A[M,K] into WMMA A-fragment layout (16-bit A 16x32, §7.12.2):
// lane<16 holds K 0..15 (v0..7); lane>=16 holds K 16..31 (contiguous k, same as B pack).
// Optional f32 add (q = src+pos), optional f16 source (FFN hidden).
// ---------------------------------------------------------------------------
__global__ void pack_a_kernel(const float* __restrict__ Af,
                              const float* __restrict__ Aadd,
                              const _Float16* __restrict__ Ah,
                              uint32_t* __restrict__ dst,
                              int M, int K) {
    int t = blockIdx.x * blockDim.x + threadIdx.x;
    int total = (M >> 4) * (K >> 5) * 256;
    if (t >= total) return;
    int lane = t & 31;
    int i    = (t >> 5) & 7;
    int tile = t >> 8;
    int ktiles = K >> 5;
    int kt = tile % ktiles;
    int mt = tile / ktiles;
    int row = (mt << 4) + (lane & 15);
    int kc  = (kt << 5) + ((lane >> 4) << 4) + (i << 1);
    float v0, v1;
    if (Ah) {
        v0 = (float)Ah[(size_t)row * K + kc];
        v1 = (float)Ah[(size_t)row * K + kc + 1];
    } else {
        v0 = Af[(size_t)row * K + kc];
        v1 = Af[(size_t)row * K + kc + 1];
        if (Aadd) {
            v0 += Aadd[(size_t)row * K + kc];
            v1 += Aadd[(size_t)row * K + kc + 1];
        }
    }
    PackPair p;
    p.hh[0] = (_Float16)v0;
    p.hh[1] = (_Float16)v1;
    dst[(size_t)tile * 256 + lane * 8 + i] = p.uu;
}

// ---------------------------------------------------------------------------
// WMMA GEMM: C[M,N] = A[M,K] * B[K,N] + bias. 8 waves/block; wave w owns a
// 16x64 tile (one A fragment x 4 B fragments -> 4 v_wmma per 32-wide K step).
// B tiles (4 KB per K-step, shared by all 8 waves) are staged into LDS with
// gfx1250 async copies (global_load_async_to_lds_b128, ASYNCcnt-tracked),
// double-buffered so the copy of K-tile k+1 overlaps the WMMAs of tile k.
// The LDS offset passed to the async op is derived from the real address of
// bsm (ptrtoint escape) so the compiler cannot fold the ds_loads to undef.
// ---------------------------------------------------------------------------
__global__ __launch_bounds__(256) void wmma_gemm_kernel(
    const uint32_t* __restrict__ Ap, const uint32_t* __restrict__ Bp,
    const float* __restrict__ bias,
    float* __restrict__ Cf, _Float16* __restrict__ Ch,
    int M, int N, int K, int relu) {

    __shared__ uint32_t bsm[2 * 1024];   // 2 buffers x 4KB (4 B-tiles each)

    int tid  = threadIdx.x;
    int lane = tid & 31;
    int wave = tid >> 5;
    int mt   = blockIdx.x * 8 + wave;
    int nt0  = blockIdx.y * 4;
    int ktiles = K >> 5;
    int ntiles = N >> 4;

    // Stage the 4 B-fragment tiles of K-tile kt into LDS buffer `buf`
    // (256 threads x 16 B = 4 KB, one async b128 per thread).
    auto stage_b = [&](int kt, int buf) {
        const uint32_t* g = Bp + ((size_t)kt * ntiles + nt0) * 256 + tid * 4;
        // Flat LDS addresses carry the aperture in [63:32]; low 32 bits are the
        // LDS byte offset (ISA §10.2). The ptrtoint also makes bsm escape.
        uint32_t ldsOff = (uint32_t)(uintptr_t)(&bsm[buf * 1024 + tid * 4]);
        asm volatile("global_load_async_to_lds_b128 %0, %1, off"
                     :: "v"(ldsOff), "v"(g)
                     : "memory");
    };

    auto load_b_frags = [&](int buf, Frag16* bf) {
        const uint32_t* bl = &bsm[buf * 1024 + lane * 8];
#pragma unroll
        for (int j = 0; j < 4; ++j) {
            *(uint4*)&bf[j].u[0] = *(const uint4*)(bl + j * 256);
            *(uint4*)&bf[j].u[4] = *(const uint4*)(bl + j * 256 + 4);
        }
    };

    stage_b(0, 0);

    v8f acc[4] = {};
    Frag16 a_cur, a_nxt, bf[4];

    const uint32_t* ap = Ap + ((size_t)mt * ktiles) * 256 + lane * 8;
    *(uint4*)&a_cur.u[0] = *(const uint4*)(ap);
    *(uint4*)&a_cur.u[4] = *(const uint4*)(ap + 4);
    ap += 256;

    for (int kt = 0; kt < ktiles - 1; ++kt) {
        stage_b(kt + 1, (kt + 1) & 1);
        *(uint4*)&a_nxt.u[0] = *(const uint4*)(ap);
        *(uint4*)&a_nxt.u[4] = *(const uint4*)(ap + 4);
        ap += 256;
        // own copy of tile kt done; copy of kt+1 may remain in flight
        asm volatile("s_wait_asynccnt 0x1" ::: "memory");
        __syncthreads();   // all waves' copies of buf[kt&1] complete

        load_b_frags(kt & 1, bf);
        __syncthreads();   // fragments in regs; buffer may be overwritten next

#pragma unroll
        for (int j = 0; j < 4; ++j) {
            acc[j] = __builtin_amdgcn_wmma_f32_16x16x32_f16(
                false, a_cur.h, false, bf[j].h, (short)0, acc[j], false, false);
        }
#pragma unroll
        for (int i = 0; i < 8; ++i) a_cur.u[i] = a_nxt.u[i];
    }

    // Last K-tile (peeled: no next copy, no trailing barrier needed).
    asm volatile("s_wait_asynccnt 0x0" ::: "memory");
    __syncthreads();
    load_b_frags((ktiles - 1) & 1, bf);
#pragma unroll
    for (int j = 0; j < 4; ++j) {
        acc[j] = __builtin_amdgcn_wmma_f32_16x16x32_f16(
            false, a_cur.h, false, bf[j].h, (short)0, acc[j], false, false);
    }

    // C/D layout: lanes 0-15 -> rows r (VGPR r), lanes 16-31 -> rows r+8.
    int rowbase = (mt << 4) + ((lane >> 4) << 3);
    int col0    = (nt0 << 4) + (lane & 15);
#pragma unroll
    for (int j = 0; j < 4; ++j) {
        int col = col0 + j * 16;
        float bv = bias ? bias[col] : 0.0f;
#pragma unroll
        for (int r = 0; r < 8; ++r) {
            float v = acc[j][r] + bv;
            if (relu) v = fmaxf(v, 0.0f);
            size_t off = (size_t)(rowbase + r) * N + col;
            if (Ch) Ch[off] = (_Float16)v;
            else    Cf[off] = v;
        }
    }
}

// ---------------------------------------------------------------------------
// Per-(token, head) softmax over 16 (level,point) logits + bilinear sampling.
// One wave per head; lane = channel d in DH=32. Corner gathers are coalesced
// 128B reads of value[b, start+idx, h, :].
// ---------------------------------------------------------------------------
__device__ __forceinline__ float fetch_corner(const float* __restrict__ vb,
                                              int xi, int yi, int H, int W) {
    bool valid = (xi >= 0) & (xi < W) & (yi >= 0) & (yi < H);
    int xc = min(max(xi, 0), W - 1);
    int yc = min(max(yi, 0), H - 1);
    float v = vb[(size_t)(yc * W + xc) * D_];
    return valid ? v : 0.0f;
}

__global__ __launch_bounds__(256) void sample_kernel(
    const float* __restrict__ value, const float* __restrict__ offs,
    const float* __restrict__ attnLog, const float* __restrict__ refpts,
    float* __restrict__ accOut) {
    int t = blockIdx.x;                 // token in [0, BS)
    int h = threadIdx.x >> 5;           // head
    int d = threadIdx.x & 31;           // channel
    int bb = t / S_TOT;

    const float* ap = attnLog + (size_t)t * (NH_ * NL_ * NP_) + h * 16;
    float logit[16];
    float mx = -1e30f;
#pragma unroll
    for (int k = 0; k < 16; ++k) { logit[k] = ap[k]; mx = fmaxf(mx, logit[k]); }
    float wsum = 0.0f;
    float w[16];
#pragma unroll
    for (int k = 0; k < 16; ++k) { w[k] = __expf(logit[k] - mx); wsum += w[k]; }
    float inv = 1.0f / wsum;

    const float* op = offs + (size_t)t * 256 + h * 32;
    const float* rp = refpts + (size_t)t * (NL_ * 2);

    const int Hs[4]     = {128, 64, 32, 16};
    const int Ws[4]     = {128, 64, 32, 16};
    const int starts[4] = {0, 16384, 20480, 21504};

    float acc = 0.0f;
#pragma unroll
    for (int l = 0; l < NL_; ++l) {
        int H = Hs[l], W = Ws[l], st = starts[l];
        float rx = rp[l * 2 + 0];
        float ry = rp[l * 2 + 1];
        const float* vbase = value + ((size_t)(bb * S_TOT + st)) * D_ + h * 32 + d;
        float invW = 1.0f / (float)W;
        float invH = 1.0f / (float)H;
#pragma unroll
        for (int p = 0; p < NP_; ++p) {
            int k = l * 4 + p;
            float lx = rx + op[k * 2 + 0] * invW;
            float ly = ry + op[k * 2 + 1] * invH;
            float x = lx * (float)W - 0.5f;
            float y = ly * (float)H - 0.5f;
            float x0f = floorf(x), y0f = floorf(y);
            float wx = x - x0f, wy = y - y0f;
            int x0 = (int)x0f, y0 = (int)y0f;
            float aw = w[k] * inv;
            float v00 = fetch_corner(vbase, x0,     y0,     H, W);
            float v01 = fetch_corner(vbase, x0 + 1, y0,     H, W);
            float v10 = fetch_corner(vbase, x0,     y0 + 1, H, W);
            float v11 = fetch_corner(vbase, x0 + 1, y0 + 1, H, W);
            acc += aw * (v00 * (1.0f - wx) * (1.0f - wy) +
                         v01 * wx * (1.0f - wy) +
                         v10 * (1.0f - wx) * wy +
                         v11 * wx * wy);
        }
    }
    accOut[(size_t)t * D_ + h * 32 + d] = acc;
}

// ---------------------------------------------------------------------------
// out = LayerNorm(A + Bv) * g + b ; one 256-thread block per token row.
// ---------------------------------------------------------------------------
__global__ __launch_bounds__(256) void add_ln_kernel(
    const float* __restrict__ A, const float* __restrict__ Bv,
    const float* __restrict__ g, const float* __restrict__ be,
    float* __restrict__ out) {
    __shared__ float sm[16];
    int t = blockIdx.x;
    int i = threadIdx.x;
    float h = A[(size_t)t * D_ + i] + Bv[(size_t)t * D_ + i];
    float s = h, s2 = h * h;
#pragma unroll
    for (int o = 16; o > 0; o >>= 1) {
        s  += __shfl_xor(s,  o, 32);
        s2 += __shfl_xor(s2, o, 32);
    }
    int wv = i >> 5, ln = i & 31;
    if (ln == 0) { sm[wv] = s; sm[8 + wv] = s2; }
    __syncthreads();
    if (i == 0) {
        float ts = 0.0f, ts2 = 0.0f;
#pragma unroll
        for (int k = 0; k < 8; ++k) { ts += sm[k]; ts2 += sm[8 + k]; }
        sm[0] = ts; sm[8] = ts2;
    }
    __syncthreads();
    float mean = sm[0] * (1.0f / D_);
    float var  = sm[8] * (1.0f / D_) - mean * mean;
    float r = rsqrtf(var + 1e-5f);
    out[(size_t)t * D_ + i] = (h - mean) * r * g[i] + be[i];
}

// ---------------------------------------------------------------------------
extern "C" void kernel_launch(void* const* d_in, const int* in_sizes, int n_in,
                              void* d_out, int out_size, void* d_ws, size_t ws_size,
                              hipStream_t stream) {
    const float* src    = (const float*)d_in[0];
    const float* pos    = (const float*)d_in[1];
    const float* refpts = (const float*)d_in[2];
    // d_in[3] spatial_shapes, d_in[4] level_start_index: compile-time constants here.
    const float* W_off  = (const float*)d_in[5];
    const float* b_off  = (const float*)d_in[6];
    const float* W_attn = (const float*)d_in[7];
    const float* b_attn = (const float*)d_in[8];
    const float* W_val  = (const float*)d_in[9];
    const float* b_val  = (const float*)d_in[10];
    const float* W_out  = (const float*)d_in[11];
    const float* b_out  = (const float*)d_in[12];
    const float* W_ffn1 = (const float*)d_in[13];
    const float* b_ffn1 = (const float*)d_in[14];
    const float* W_ffn2 = (const float*)d_in[15];
    const float* b_ffn2 = (const float*)d_in[16];
    const float* ln1_g  = (const float*)d_in[17];
    const float* ln1_b  = (const float*)d_in[18];
    const float* ln2_g  = (const float*)d_in[19];
    const float* ln2_b  = (const float*)d_in[20];

    char* ws = (char*)d_ws;
    const size_t U = (size_t)BS_ * D_ * sizeof(float);   // 44,564,480 B

    // Packed weights (f16 fragments) at front of workspace.
    uint32_t* WvalP  = (uint32_t*)(ws + 0);
    uint32_t* WoffP  = (uint32_t*)(ws + 131072);
    uint32_t* WattnP = (uint32_t*)(ws + 262144);
    uint32_t* WoutP  = (uint32_t*)(ws + 327680);
    uint32_t* Wf1P   = (uint32_t*)(ws + 458752);
    uint32_t* Wf2P   = (uint32_t*)(ws + 983040);
    const size_t base = (size_t)2 * 1024 * 1024;

    uint32_t*  srcPack = (uint32_t*)(ws + base);                 // U/2
    uint32_t*  qPack   = (uint32_t*)(ws + base + U / 2);         // U/2
    float*     valueF  = (float*)(ws + base + U);                // U
    float*     offF    = (float*)(ws + base + 2 * U);            // U
    float*     attnF   = (float*)(ws + base + 3 * U);            // U/2
    _Float16*  h1F16   = (_Float16*)(ws + base + 3 * U + U / 2); // 2U
    uint32_t*  h1Pack  = (uint32_t*)(ws + base + 5 * U + U / 2); // 2U
    // Aliased (dead-buffer reuse, strictly sequential writes after last read):
    float*     accF    = (float*)srcPack;   // U  (src/q packs dead)
    uint32_t*  accPack = (uint32_t*)offF;   // U/2 (off dead after sampling)
    float*     src2F   = valueF;            // U  (value dead after sampling)
    float*     xF      = offF;              // U  (accPack dead after out-GEMM)
    uint32_t*  xPack   = (uint32_t*)attnF;  // U/2 (attn dead after sampling)
    float*     ffn2F   = valueF;            // U  (src2 dead after LN1)
    (void)in_sizes; (void)n_in; (void)out_size; (void)ws_size;

    // --- Pack weights ---
    pack_b_kernel<<<128, 256, 0, stream>>>(W_val,  WvalP,  256, 256);
    pack_b_kernel<<<128, 256, 0, stream>>>(W_off,  WoffP,  256, 256);
    pack_b_kernel<<< 64, 256, 0, stream>>>(W_attn, WattnP, 256, 128);
    pack_b_kernel<<<128, 256, 0, stream>>>(W_out,  WoutP,  256, 256);
    pack_b_kernel<<<512, 256, 0, stream>>>(W_ffn1, Wf1P,   256, 1024);
    pack_b_kernel<<<512, 256, 0, stream>>>(W_ffn2, Wf2P,   1024, 256);

    // --- Pack activations: src and q = src + pos ---
    pack_a_kernel<<<21760, 256, 0, stream>>>(src, nullptr, nullptr, srcPack, BS_, 256);
    pack_a_kernel<<<21760, 256, 0, stream>>>(src, pos,     nullptr, qPack,   BS_, 256);

    // --- value / offsets / attention logits GEMMs ---
    wmma_gemm_kernel<<<dim3(340, 4), 256, 0, stream>>>(srcPack, WvalP,  b_val,  valueF, nullptr, BS_, 256,  256, 0);
    wmma_gemm_kernel<<<dim3(340, 4), 256, 0, stream>>>(qPack,   WoffP,  b_off,  offF,   nullptr, BS_, 256,  256, 0);
    wmma_gemm_kernel<<<dim3(340, 2), 256, 0, stream>>>(qPack,   WattnP, b_attn, attnF,  nullptr, BS_, 128,  256, 0);

    // --- softmax + bilinear sampling ---
    sample_kernel<<<BS_, 256, 0, stream>>>(valueF, offF, attnF, refpts, accF);

    // --- output projection + LN1 ---
    pack_a_kernel<<<21760, 256, 0, stream>>>(accF, nullptr, nullptr, accPack, BS_, 256);
    wmma_gemm_kernel<<<dim3(340, 4), 256, 0, stream>>>(accPack, WoutP, b_out, src2F, nullptr, BS_, 256, 256, 0);
    add_ln_kernel<<<BS_, 256, 0, stream>>>(src, src2F, ln1_g, ln1_b, xF);

    // --- FFN: GEMM1 (+ReLU -> f16) -> GEMM2 -> residual LN2 ---
    pack_a_kernel<<<21760, 256, 0, stream>>>(xF, nullptr, nullptr, xPack, BS_, 256);
    wmma_gemm_kernel<<<dim3(340, 16), 256, 0, stream>>>(xPack, Wf1P, b_ffn1, nullptr, h1F16, BS_, 1024, 256, 1);
    pack_a_kernel<<<87040, 256, 0, stream>>>(nullptr, nullptr, h1F16, h1Pack, BS_, 1024);
    wmma_gemm_kernel<<<dim3(340, 4), 256, 0, stream>>>(h1Pack, Wf2P, b_ffn2, ffn2F, nullptr, BS_, 256, 1024, 0);
    add_ln_kernel<<<BS_, 256, 0, stream>>>(xF, ffn2F, ln2_g, ln2_b, (float*)d_out);
}
